// InteractionBlock_11510512353346
// MI455X (gfx1250) — compile-verified, matrix operand order checked by reading, not run on hardware
//
#include <hip/hip_runtime.h>
#include <hip/hip_bf16.h>

// ---------------------------------------------------------------------------
// PhysNet InteractionBlock for gfx1250 (MI455X), bf16 WMMA path.
// ---------------------------------------------------------------------------

typedef __bf16 bf16_t;
typedef __attribute__((ext_vector_type(16))) __bf16 v16bf;
typedef __attribute__((ext_vector_type(8)))  float  v8f;

#define FDIM 128          // feature dim
#define KDIM 64           // radial basis dim
#define LDW  140          // LDS stride (halfs) for 128-wide rows: 70 dwords -> conflict-free frags
#define LDWE 76           // LDS stride (halfs) for 64-wide rows:  38 dwords -> conflict-free frags
#define LN2   0.6931471805599453f
#define LOG2E 1.4426950408889634f

// shifted softplus: softplus(x) - ln2, numerically stable; ssp(0)==0.
// Uses raw v_exp_f32/v_log_f32 (base-2): the log argument 1+exp2(-|x|*log2e)
// lies in [1,2] so no denormal/range fixups are needed.
__device__ __forceinline__ float ssp(float x) {
  float t = __builtin_amdgcn_exp2f(-fabsf(x) * LOG2E);
  return fmaxf(x, 0.f) + LN2 * (__builtin_amdgcn_logf(1.f + t) - 1.f);
}

// K index of pair v within a 32-wide k-chunk, per CDNA5 16-bit A/B fragment layout:
// lanes 0-15 hold K 0..7 (v0..3) and 16..23 (v4..7); lanes 16-31 hold K 8..15 and 24..31.
__device__ __forceinline__ int kbase_of(int v, int koff) {
  return koff + 2 * v + ((v >= 4) ? 8 : 0);
}

// Load a 16x32 bf16 fragment from row-major fp32 global memory (row = this lane's M/N).
__device__ __forceinline__ v16bf load_frag_g(const float* __restrict__ src, int ld,
                                             int row, int kc, bool do_ssp, bool valid) {
  int koff = ((threadIdx.x >> 4) & 1) * 8;
  const float* p = src + (size_t)row * ld + kc * 32;
  v16bf a;
#pragma unroll
  for (int v = 0; v < 8; ++v) {
    int kb = kbase_of(v, koff);
    float x0 = 0.f, x1 = 0.f;
    if (valid) { float2 f = *(const float2*)(p + kb); x0 = f.x; x1 = f.y; }
    if (do_ssp) { x0 = ssp(x0); x1 = ssp(x1); }   // ssp(0)==0, so invalid rows stay zero
    a[2 * v] = (bf16_t)x0;
    a[2 * v + 1] = (bf16_t)x1;
  }
  return a;
}

// Load a 16x32 bf16 fragment from LDS (rc = this lane's row/col index into the tile).
__device__ __forceinline__ v16bf load_frag_lds(const bf16_t* base, int ld, int rc, int kc) {
  int koff = ((threadIdx.x >> 4) & 1) * 8;
  const bf16_t* p = base + rc * ld + kc * 32;
  v16bf a;
#pragma unroll
  for (int v = 0; v < 8; ++v) {
    int kb = kbase_of(v, koff);
    a[2 * v] = p[kb];
    a[2 * v + 1] = p[kb + 1];
  }
  return a;
}

__device__ __forceinline__ v8f vzero() {
  v8f z = {0.f, 0.f, 0.f, 0.f, 0.f, 0.f, 0.f, 0.f};
  return z;
}

// ---------------------------------------------------------------------------
// k_dense: out = [ssp]( [ssp](in) @ W^T + b )  [+ u[col]*addsrc]
// Block: 256 threads = 8 waves; 64 rows x 128 cols per block.
// Wave w: row-tile w>>1 (16 rows), column half w&1 (4 col-tiles of 16).
// ---------------------------------------------------------------------------
__global__ __launch_bounds__(256)
void k_dense(const float* in, const float* __restrict__ W, const float* __restrict__ b,
             float* out, int n, int sspIn, int sspOut,
             const float* __restrict__ uvec, const float* __restrict__ addsrc) {
  __shared__ bf16_t sW[FDIM * LDW];
  for (int i = threadIdx.x; i < FDIM * FDIM; i += 256) {
    int r = i >> 7, c = i & 127;
    sW[r * LDW + c] = (bf16_t)W[i];
  }
  __syncthreads();

  int lane = threadIdx.x & 31;
  int wave = threadIdx.x >> 5;
  int rowtile = wave >> 1;
  int colhalf = wave & 1;
  int row0 = blockIdx.x * 64 + rowtile * 16;
  int myrow = row0 + (lane & 15);
  bool valid = myrow < n;

  v8f acc[4];
#pragma unroll
  for (int ct = 0; ct < 4; ++ct) acc[ct] = vzero();

#pragma unroll
  for (int kc = 0; kc < 4; ++kc) {
    v16bf a = load_frag_g(in, FDIM, myrow, kc, sspIn != 0, valid);
#pragma unroll
    for (int ct = 0; ct < 4; ++ct) {
      int col = colhalf * 64 + ct * 16 + (lane & 15);
      v16bf bf = load_frag_lds(sW, LDW, col, kc);  // B[k][col] = W[col][k]
      acc[ct] = __builtin_amdgcn_wmma_f32_16x16x32_bf16(
          false, a, false, bf, (short)0, acc[ct], false, false);
    }
  }

  __syncthreads();  // make in-place (out==in) safe: all A-loads of this block done

  int mbase = (lane >> 4) * 8;  // D layout: lanes 16-31 hold rows 8..15
  int ncol = lane & 15;
#pragma unroll
  for (int ct = 0; ct < 4; ++ct) {
    int col = colhalf * 64 + ct * 16 + ncol;
    float bias = b[col];
    float uc = uvec ? uvec[col] : 0.f;
#pragma unroll
    for (int v = 0; v < 8; ++v) {
      int r = row0 + mbase + v;
      if (r < n) {
        float val = acc[ct][v] + bias;
        if (sspOut) val = ssp(val);
        if (addsrc) val += uc * addsrc[(size_t)r * FDIM + col];
        out[(size_t)r * FDIM + col] = val;
      }
    }
  }
}

// ---------------------------------------------------------------------------
// k_dense2: two GEMMs sharing the same activated A operand:
//   outa = ssp( ssp(in) @ Wa^T + ba ),  outb = ssp( ssp(in) @ Wb^T + bb )
// Halves the feature reads + ssp work for the xi / xj_all stage.
// Dynamic LDS: two weight tiles.
// ---------------------------------------------------------------------------
__global__ __launch_bounds__(256)
void k_dense2(const float* __restrict__ in,
              const float* __restrict__ Wa, const float* __restrict__ ba,
              const float* __restrict__ Wb, const float* __restrict__ bb,
              float* __restrict__ outa, float* __restrict__ outb, int n) {
  extern __shared__ char smem[];
  bf16_t* sWa = (bf16_t*)smem;
  bf16_t* sWb = sWa + FDIM * LDW;
  for (int i = threadIdx.x; i < FDIM * FDIM; i += 256) {
    int r = i >> 7, c = i & 127;
    sWa[r * LDW + c] = (bf16_t)Wa[i];
    sWb[r * LDW + c] = (bf16_t)Wb[i];
  }
  __syncthreads();

  int lane = threadIdx.x & 31;
  int wave = threadIdx.x >> 5;
  int rowtile = wave >> 1;
  int colhalf = wave & 1;
  int row0 = blockIdx.x * 64 + rowtile * 16;
  int myrow = row0 + (lane & 15);
  bool valid = myrow < n;

  v8f acca[4], accb[4];
#pragma unroll
  for (int ct = 0; ct < 4; ++ct) { acca[ct] = vzero(); accb[ct] = vzero(); }

#pragma unroll
  for (int kc = 0; kc < 4; ++kc) {
    v16bf a = load_frag_g(in, FDIM, myrow, kc, true, valid);
#pragma unroll
    for (int ct = 0; ct < 4; ++ct) {
      int col = colhalf * 64 + ct * 16 + (lane & 15);
      v16bf bfa = load_frag_lds(sWa, LDW, col, kc);
      acca[ct] = __builtin_amdgcn_wmma_f32_16x16x32_bf16(
          false, a, false, bfa, (short)0, acca[ct], false, false);
      v16bf bfb = load_frag_lds(sWb, LDW, col, kc);
      accb[ct] = __builtin_amdgcn_wmma_f32_16x16x32_bf16(
          false, a, false, bfb, (short)0, accb[ct], false, false);
    }
  }

  int mbase = (lane >> 4) * 8;
  int ncol = lane & 15;
#pragma unroll
  for (int ct = 0; ct < 4; ++ct) {
    int col = colhalf * 64 + ct * 16 + ncol;
    float biasa = ba[col];
    float biasb = bb[col];
#pragma unroll
    for (int v = 0; v < 8; ++v) {
      int r = row0 + mbase + v;
      if (r < n) {
        size_t off = (size_t)r * FDIM + col;
        outa[off] = ssp(acca[ct][v] + biasa);
        outb[off] = ssp(accb[ct][v] + biasb);
      }
    }
  }
}

// ---------------------------------------------------------------------------
// k_residual: out = m + ssp( ssp(m) @ W1^T + b1 ) @ W2^T + b2   (fused, one pass)
// Intermediate h tile staged in LDS as bf16 between the two WMMA passes.
// Dynamic LDS: 2 weight tiles + 64-row h tile.
// ---------------------------------------------------------------------------
__global__ __launch_bounds__(256)
void k_residual(const float* m, const float* __restrict__ W1, const float* __restrict__ b1,
                const float* __restrict__ W2, const float* __restrict__ b2,
                float* out, int n) {
  extern __shared__ char smem[];
  bf16_t* sW1 = (bf16_t*)smem;
  bf16_t* sW2 = sW1 + FDIM * LDW;
  bf16_t* sH  = sW2 + FDIM * LDW;   // 64 x LDW

  for (int i = threadIdx.x; i < FDIM * FDIM; i += 256) {
    int r = i >> 7, c = i & 127;
    sW1[r * LDW + c] = (bf16_t)W1[i];
    sW2[r * LDW + c] = (bf16_t)W2[i];
  }
  __syncthreads();

  int lane = threadIdx.x & 31;
  int wave = threadIdx.x >> 5;
  int rowtile = wave >> 1;
  int colhalf = wave & 1;
  int row0 = blockIdx.x * 64 + rowtile * 16;
  int myrow = row0 + (lane & 15);
  bool valid = myrow < n;
  int mbase = (lane >> 4) * 8;
  int ncol = lane & 15;

  // ---- GEMM1: h = ssp( ssp(m) @ W1^T + b1 ) ----
  v8f acc[4];
#pragma unroll
  for (int ct = 0; ct < 4; ++ct) acc[ct] = vzero();
#pragma unroll
  for (int kc = 0; kc < 4; ++kc) {
    v16bf a = load_frag_g(m, FDIM, myrow, kc, true, valid);
#pragma unroll
    for (int ct = 0; ct < 4; ++ct) {
      int col = colhalf * 64 + ct * 16 + (lane & 15);
      v16bf bf = load_frag_lds(sW1, LDW, col, kc);
      acc[ct] = __builtin_amdgcn_wmma_f32_16x16x32_bf16(
          false, a, false, bf, (short)0, acc[ct], false, false);
    }
  }
#pragma unroll
  for (int ct = 0; ct < 4; ++ct) {
    int col = colhalf * 64 + ct * 16 + ncol;
    float bias = b1[col];
#pragma unroll
    for (int v = 0; v < 8; ++v) {
      int lrow = rowtile * 16 + mbase + v;     // row within 64-row tile
      sH[lrow * LDW + col] = (bf16_t)ssp(acc[ct][v] + bias);
    }
  }
  __syncthreads();

  // ---- GEMM2: out = m + h @ W2^T + b2 ----
#pragma unroll
  for (int ct = 0; ct < 4; ++ct) acc[ct] = vzero();
#pragma unroll
  for (int kc = 0; kc < 4; ++kc) {
    v16bf a = load_frag_lds(sH + rowtile * 16 * LDW, LDW, lane & 15, kc);
#pragma unroll
    for (int ct = 0; ct < 4; ++ct) {
      int col = colhalf * 64 + ct * 16 + (lane & 15);
      v16bf bf = load_frag_lds(sW2, LDW, col, kc);
      acc[ct] = __builtin_amdgcn_wmma_f32_16x16x32_bf16(
          false, a, false, bf, (short)0, acc[ct], false, false);
    }
  }
#pragma unroll
  for (int ct = 0; ct < 4; ++ct) {
    int col = colhalf * 64 + ct * 16 + ncol;
    float bias = b2[col];
#pragma unroll
    for (int v = 0; v < 8; ++v) {
      int r = row0 + mbase + v;
      if (r < n) {
        size_t off = (size_t)r * FDIM + col;
        out[off] = m[off] + acc[ct][v] + bias;   // element-disjoint: in-place safe
      }
    }
  }
}

// ---------------------------------------------------------------------------
// k_edge: g = desc @ Wdesc^T (WMMA, K=64); pxj = g * xja[idx_j];
//         msg[idx_i] += pxj via run-length-compressed float atomics.
// Block: 64 edges. idx_i is sorted, each lane owns 8 consecutive edge rows
// of one column -> merge equal-index runs in registers before the atomic.
// ---------------------------------------------------------------------------
__global__ __launch_bounds__(256)
void k_edge(const float* __restrict__ desc, const float* __restrict__ Wdesc,
            const int* __restrict__ idx_i, const int* __restrict__ idx_j,
            const float* __restrict__ xja, float* msg, int E) {
  __shared__ bf16_t sW[FDIM * LDWE];
  __shared__ int sI[64], sJ[64];
  for (int i = threadIdx.x; i < FDIM * KDIM; i += 256) {
    int r = i >> 6, c = i & 63;
    sW[r * LDWE + c] = (bf16_t)Wdesc[i];
  }
  int e0 = blockIdx.x * 64;
  if (threadIdx.x < 64) {
    int e = e0 + threadIdx.x;
    sI[threadIdx.x] = (e < E) ? idx_i[e] : 0;
    sJ[threadIdx.x] = (e < E) ? idx_j[e] : 0;
  }
  __syncthreads();

  int lane = threadIdx.x & 31;
  int wave = threadIdx.x >> 5;
  int rowtile = wave >> 1;
  int colhalf = wave & 1;
  int myedge = e0 + rowtile * 16 + (lane & 15);
  bool valid = myedge < E;

  v8f acc[4];
#pragma unroll
  for (int ct = 0; ct < 4; ++ct) acc[ct] = vzero();
#pragma unroll
  for (int kc = 0; kc < 2; ++kc) {
    v16bf a = load_frag_g(desc, KDIM, myedge, kc, false, valid);
#pragma unroll
    for (int ct = 0; ct < 4; ++ct) {
      int col = colhalf * 64 + ct * 16 + (lane & 15);
      v16bf bf = load_frag_lds(sW, LDWE, col, kc);
      acc[ct] = __builtin_amdgcn_wmma_f32_16x16x32_bf16(
          false, a, false, bf, (short)0, acc[ct], false, false);
    }
  }

  int mbase = (lane >> 4) * 8;
  int ncol = lane & 15;
#pragma unroll
  for (int ct = 0; ct < 4; ++ct) {
    int col = colhalf * 64 + ct * 16 + ncol;
    float sum = 0.f;
    int cur = -1;
#pragma unroll
    for (int v = 0; v < 8; ++v) {
      int le = rowtile * 16 + mbase + v;       // local edge 0..63, consecutive per lane
      if (e0 + le < E) {
        int ii = sI[le], jj = sJ[le];
        float val = acc[ct][v] * xja[(size_t)jj * FDIM + col];
        if (ii != cur) {
          if (cur >= 0) atomicAdd(&msg[(size_t)cur * FDIM + col], sum);
          cur = ii;
          sum = val;
        } else {
          sum += val;
        }
      }
    }
    if (cur >= 0) atomicAdd(&msg[(size_t)cur * FDIM + col], sum);
  }
}

// ---------------------------------------------------------------------------
// Host-side launch sequence (graph-capture safe; everything on `stream`).
// ---------------------------------------------------------------------------
extern "C" void kernel_launch(void* const* d_in, const int* in_sizes, int n_in,
                              void* d_out, int out_size, void* d_ws, size_t ws_size,
                              hipStream_t stream) {
  const float* features = (const float*)d_in[0];
  const float* desc     = (const float*)d_in[1];
  const int*   idx_i    = (const int*)d_in[2];
  const int*   idx_j    = (const int*)d_in[3];
  const float* Wdesc    = (const float*)d_in[4];
  const float* Wi   = (const float*)d_in[5];
  const float* bi   = (const float*)d_in[6];
  const float* Wj   = (const float*)d_in[7];
  const float* bj   = (const float*)d_in[8];
  const float* Wri1 = (const float*)d_in[9];
  const float* bri1 = (const float*)d_in[10];
  const float* Wri2 = (const float*)d_in[11];
  const float* bri2 = (const float*)d_in[12];
  const float* Wd   = (const float*)d_in[13];
  const float* bd   = (const float*)d_in[14];
  const float* u    = (const float*)d_in[15];
  const float* Wra1 = (const float*)d_in[16];
  const float* bra1 = (const float*)d_in[17];
  const float* Wra2 = (const float*)d_in[18];
  const float* bra2 = (const float*)d_in[19];

  const int N = in_sizes[0] / FDIM;
  const int E = in_sizes[2];
  const int NRI = 3, NRA = 2;

  float* msg = (float*)d_ws;                 // xi -> message -> x (in place)
  float* xja = msg + (size_t)N * FDIM;       // xj_all (free after k_edge)
  float* out = (float*)d_out;

  dim3 blk(256);
  dim3 gnode((unsigned)((N + 63) / 64));
  dim3 gedge((unsigned)((E + 63) / 64));
  const size_t d2Lds  = (size_t)(2 * FDIM * LDW) * sizeof(bf16_t);
  const size_t resLds = (size_t)(2 * FDIM * LDW + 64 * LDW) * sizeof(bf16_t);

  // xi = ssp(ssp(x)@Wi^T+bi) -> msg ; xj_all = ssp(ssp(x)@Wj^T+bj) -> xja
  // (one pass over features; fully overwrites msg: no zero-init needed)
  k_dense2<<<gnode, blk, d2Lds, stream>>>(features, Wi, bi, Wj, bj, msg, xja, N);
  // msg += segment_sum(g * xja[idx_j])  => msg = message
  k_edge<<<gedge, blk, 0, stream>>>(desc, Wdesc, idx_i, idx_j, xja, msg, E);
  // interaction residual stack (in place)
  for (int k = 0; k < NRI; ++k)
    k_residual<<<gnode, blk, resLds, stream>>>(msg, Wri1 + (size_t)k * FDIM * FDIM,
                                               bri1 + (size_t)k * FDIM,
                                               Wri2 + (size_t)k * FDIM * FDIM,
                                               bri2 + (size_t)k * FDIM, msg, N);
  // x = u * features + ssp(message) @ Wd^T + bd   (in place on msg)
  k_dense<<<gnode, blk, 0, stream>>>(msg, Wd, bd, msg, N, 1, 0, u, features);
  // atomic residual stack; last layer writes d_out
  for (int k = 0; k < NRA; ++k) {
    float* dst = (k == NRA - 1) ? out : msg;
    k_residual<<<gnode, blk, resLds, stream>>>(msg, Wra1 + (size_t)k * FDIM * FDIM,
                                               bra1 + (size_t)k * FDIM,
                                               Wra2 + (size_t)k * FDIM * FDIM,
                                               bra2 + (size_t)k * FDIM, dst, N);
  }
}
